// LSTM_6957847019793
// MI455X (gfx1250) — matrix operand order repaired and checked by reading, not compiled, not run on hardware
//
#include <hip/hip_runtime.h>
#include <hip/hip_bf16.h>

#define BB   128
#define TT   1024
#define II   256
#define HH   512
#define KK   768   // I + H
#define OO   256
#define NBLK 64
#define NTHR 128   // 4 waves per block -> 256 waves total = 256 (16x16) tiles
#define LDW  776   // padded LDS row stride (768 + 8) -> conflict-free ds_load_b128

typedef __attribute__((ext_vector_type(16))) __bf16 v16bf;
typedef __attribute__((ext_vector_type(8)))  __bf16 v8bf;
typedef __attribute__((ext_vector_type(8)))  float  v8f;

// ---------- fragment loaders (per CDNA5 ISA 7.12.2 layouts) ----------
// A matrix 16x32 bf16: lane<16 -> row M=lane, elems = K{k0+0..7, k0+16..23}
//                      lane>=16 -> same rows, K{k0+8..15, k0+24..31}
template <typename P>
__device__ __forceinline__ v16bf load_fragA(P base, int ld, int row, int k0, int half) {
  P p = base + row * ld + k0 + half * 8;
  v8bf lo = *(const v8bf*)(p);
  v8bf hi = *(const v8bf*)(p + 16);
  v16bf r;
#pragma unroll
  for (int i = 0; i < 8; ++i) { r[i] = lo[i]; r[i + 8] = hi[i]; }
  return r;
}

// B matrix 32x16 bf16 (K x N): lane<16 -> col N=lane, K = {k0+0..7, k0+16..23}
//                              lane>=16 -> K = {k0+8..15, k0+24..31}
// We compute combined @ W^T, so "column n" of B == row n of W (contiguous).
template <typename P>
__device__ __forceinline__ v16bf load_fragB(P base, int ld, int col, int k0, int half) {
  P p = base + col * ld + k0 + half * 8;
  v8bf lo = *(const v8bf*)(p);
  v8bf hi = *(const v8bf*)(p + 16);
  v16bf r;
#pragma unroll
  for (int i = 0; i < 8; ++i) { r[i] = lo[i]; r[i + 8] = hi[i]; }
  return r;
}

__device__ __forceinline__ float fast_sigmoid(float x) {
  return 1.0f / (1.0f + __expf(-x));
}

// ---------- grid-wide split barrier (graph-capture safe, reset each launch) ----------
__device__ __forceinline__ void grid_sync(unsigned* bar, unsigned nblk) {
  __threadfence();     // make this block's global writes visible before signaling
  __syncthreads();
  if (threadIdx.x == 0) {
    unsigned* cnt = bar;
    unsigned* gen = bar + 1;
    unsigned g = __hip_atomic_load(gen, __ATOMIC_ACQUIRE, __HIP_MEMORY_SCOPE_AGENT);
    if (__hip_atomic_fetch_add(cnt, 1u, __ATOMIC_ACQ_REL, __HIP_MEMORY_SCOPE_AGENT) == nblk - 1u) {
      __hip_atomic_store(cnt, 0u, __ATOMIC_RELAXED, __HIP_MEMORY_SCOPE_AGENT);
      __hip_atomic_fetch_add(gen, 1u, __ATOMIC_ACQ_REL, __HIP_MEMORY_SCOPE_AGENT);
    } else {
      while (__hip_atomic_load(gen, __ATOMIC_ACQUIRE, __HIP_MEMORY_SCOPE_AGENT) == g) {
        __builtin_amdgcn_s_sleep(1);
      }
    }
  }
  __syncthreads();
  __threadfence();
}

// ---------- preprocessing: fp32 weights -> bf16, reset barrier ----------
__global__ void lstm_preproc(const float* __restrict__ Wf, const float* __restrict__ Wi,
                             const float* __restrict__ Wc, const float* __restrict__ Wo,
                             const float* __restrict__ Wlin_f,
                             __bf16* __restrict__ Wg, __bf16* __restrict__ Wl,
                             unsigned* __restrict__ bar) {
  int gid = blockIdx.x * blockDim.x + threadIdx.x;
  int stride = gridDim.x * blockDim.x;
  const int NW = HH * KK;
  for (int idx = gid; idx < NW; idx += stride) {
    Wg[0 * NW + idx] = (__bf16)Wf[idx];
    Wg[1 * NW + idx] = (__bf16)Wi[idx];
    Wg[2 * NW + idx] = (__bf16)Wc[idx];
    Wg[3 * NW + idx] = (__bf16)Wo[idx];
  }
  for (int idx = gid; idx < OO * HH; idx += stride) Wl[idx] = (__bf16)Wlin_f[idx];
  if (gid == 0) { bar[0] = 0u; bar[1] = 0u; }
}

// ---------- persistent LSTM scan with ping-pong activations ----------
__global__ void __launch_bounds__(NTHR, 1)
lstm_persistent(const float* __restrict__ x,          // [B,T,I]
                const float* __restrict__ b_f, const float* __restrict__ b_i,
                const float* __restrict__ b_c, const float* __restrict__ b_o,
                const float* __restrict__ b_lin,
                const __bf16* __restrict__ Wg,        // [4][H][K] bf16
                const __bf16* __restrict__ Wl,        // [O][H] bf16
                __bf16* __restrict__ comb,            // [2][B][K] bf16 ([x_t | h_t])
                unsigned* __restrict__ bar,
                float* __restrict__ out) {            // [B][O]
  extern __shared__ __bf16 sW[];                      // [4][16][LDW]

  const int tid    = threadIdx.x;
  const int wave   = tid >> 5;                        // 0..3
  const int lane   = tid & 31;
  const int lane16 = lane & 15;
  const int half   = lane >> 4;
  const int gid    = blockIdx.x * NTHR + tid;         // 0..8191
  const int GSZ    = NBLK * NTHR;                     // 8192

  // tile mapping: 8 M-tiles x 32 N-tiles over 64 blocks x 4 waves
  const int n0     = (blockIdx.x >> 1) * 16;          // block's hidden-column strip
  const int m_base = (blockIdx.x & 1) * 64 + wave * 16;
  const int n      = n0 + lane16;                     // this lane's hidden column

  // --- stage this block's weight slice (16 rows x 4 gates) into LDS ---
  for (int idx = tid; idx < 4 * 16 * KK; idx += NTHR) {
    int g = idx / (16 * KK);
    int rem = idx - g * (16 * KK);
    int r = rem / KK;
    int c = rem - r * KK;
    sW[(g * 16 + r) * LDW + c] = Wg[(size_t)g * HH * KK + (size_t)(n0 + r) * KK + c];
  }

  // --- init buffer 0: h part = 0, x part = x[:,0,:] ---
  for (int idx = gid; idx < BB * HH; idx += GSZ)
    comb[(idx >> 9) * KK + II + (idx & 511)] = (__bf16)0.0f;
  for (int idx = gid; idx < BB * II; idx += GSZ)
    comb[(idx >> 8) * KK + (idx & 255)] =
        (__bf16)x[(size_t)(idx >> 8) * TT * II + (idx & 255)];

  const float bfv = b_f[n], biv = b_i[n], bcv = b_c[n], bov = b_o[n];

  // per-lane cell state: rows m_base + half*8 + v, column n (never leaves VGPRs)
  float Creg[8];
#pragma unroll
  for (int v = 0; v < 8; ++v) Creg[v] = 0.0f;

  __syncthreads();  // LDS weights ready (block-local)

  const __bf16* sWf = sW + 0 * 16 * LDW;
  const __bf16* sWi = sW + 1 * 16 * LDW;
  const __bf16* sWc = sW + 2 * 16 * LDW;
  const __bf16* sWo = sW + 3 * 16 * LDW;
  const int rowA = m_base + lane16;

  for (int t = 0; t < TT; ++t) {
    grid_sync(bar, NBLK);   // everyone's step-t inputs committed; everyone's
                            // step-(t-1) reads finished (barrier is after them)

    const __bf16* cur = comb + (size_t)(t & 1) * BB * KK;        // [x_t | h_t]
    __bf16*       nxt = (__bf16*)comb + (size_t)((t + 1) & 1) * BB * KK;

    // stream next timestep's x into the *other* buffer first, so these stores
    // overlap the matmul below (nobody reads nxt until after the next barrier)
    if (t + 1 < TT) {
      for (int idx = gid; idx < BB * II; idx += GSZ)
        nxt[(idx >> 8) * KK + (idx & 255)] =
            (__bf16)x[(size_t)(idx >> 8) * TT * II + (size_t)(t + 1) * II + (idx & 255)];
    }

    // bias folded into accumulator init (fixed column n per lane)
    v8f aF, aI, aC, aO;
#pragma unroll
    for (int v = 0; v < 8; ++v) { aF[v] = bfv; aI[v] = biv; aC[v] = bcv; aO[v] = bov; }

    for (int k0 = 0; k0 < KK; k0 += 32) {
      v16bf a  = load_fragA(cur, KK, rowA, k0, half);
      v16bf wf = load_fragB(sWf, LDW, lane16, k0, half);
      v16bf wi = load_fragB(sWi, LDW, lane16, k0, half);
      v16bf wc = load_fragB(sWc, LDW, lane16, k0, half);
      v16bf wo = load_fragB(sWo, LDW, lane16, k0, half);
      aF = __builtin_amdgcn_wmma_f32_16x16x32_bf16(false, a, false, wf, (short)0, aF, false, false);
      aI = __builtin_amdgcn_wmma_f32_16x16x32_bf16(false, a, false, wi, (short)0, aI, false, false);
      aC = __builtin_amdgcn_wmma_f32_16x16x32_bf16(false, a, false, wc, (short)0, aC, false, false);
      aO = __builtin_amdgcn_wmma_f32_16x16x32_bf16(false, a, false, wo, (short)0, aO, false, false);
    }

    // elementwise LSTM update; D layout: lane holds (m = m_base + half*8 + v, n)
#pragma unroll
    for (int v = 0; v < 8; ++v) {
      int m = m_base + half * 8 + v;
      float fg = fast_sigmoid(aF[v]);
      float ig = fast_sigmoid(aI[v]);
      float cg = tanhf(aC[v]);
      float og = fast_sigmoid(aO[v]);
      float Cn = fg * Creg[v] + ig * cg;
      Creg[v] = Cn;
      nxt[m * KK + II + n] = (__bf16)(og * tanhf(Cn));  // h_{t+1}
    }
  }

  grid_sync(bar, NBLK);     // h_T committed (in buffer TT & 1 == 0)

  // ---- final projection: out = h_T @ W_lin^T + b_lin  (128x512 x 512x256) ----
  const __bf16* hfin = comb + (size_t)(TT & 1) * BB * KK + II;   // h part, ld = KK
  int gw = blockIdx.x * 4 + wave;     // 256 waves, need 8 x 16 = 128 tiles
  if (gw < 128) {
    int mt = gw & 7;
    int nt = gw >> 3;
    v8f acc = {};
    for (int k0 = 0; k0 < HH; k0 += 32) {
      v16bf a = load_fragA(hfin, KK, mt * 16 + lane16, k0, half);
      v16bf b = load_fragB(Wl, HH, nt * 16 + lane16, k0, half);
      acc = __builtin_amdgcn_wmma_f32_16x16x32_bf16(false, a, false, b, (short)0, acc, false, false);
    }
    int no = nt * 16 + lane16;
    float bb = b_lin[no];
#pragma unroll
    for (int v = 0; v < 8; ++v) {
      int m = mt * 16 + half * 8 + v;
      out[m * OO + no] = acc[v] + bb;
    }
  }
}

extern "C" void kernel_launch(void* const* d_in, const int* in_sizes, int n_in,
                              void* d_out, int out_size, void* d_ws, size_t ws_size,
                              hipStream_t stream) {
  const float* x    = (const float*)d_in[0];
  const float* Wf   = (const float*)d_in[1];
  const float* bf   = (const float*)d_in[2];
  const float* Wi   = (const float*)d_in[3];
  const float* bi   = (const float*)d_in[4];
  const float* Wc   = (const float*)d_in[5];
  const float* bc   = (const float*)d_in[6];
  const float* Wo   = (const float*)d_in[7];
  const float* bo   = (const float*)d_in[8];
  const float* Wlin = (const float*)d_in[9];
  const float* blin = (const float*)d_in[10];
  float* out = (float*)d_out;

  // workspace layout (bytes)
  char* ws = (char*)d_ws;
  size_t offWg   = 0;                                 // 4*512*768 bf16 = 3,145,728 B
  size_t offWl   = offWg + (size_t)4 * HH * KK * 2;   //   256*512 bf16 =   262,144 B
  size_t offComb = offWl + (size_t)OO * HH * 2;       // 2*128*768 bf16 =   393,216 B
  size_t offBar  = offComb + (size_t)2 * BB * KK * 2; //   2 x u32
  __bf16*   Wg   = (__bf16*)(ws + offWg);
  __bf16*   Wl   = (__bf16*)(ws + offWl);
  __bf16*   comb = (__bf16*)(ws + offComb);
  unsigned* bar  = (unsigned*)(ws + offBar);

  lstm_preproc<<<256, 256, 0, stream>>>(Wf, Wi, Wc, Wo, Wlin, Wg, Wl, bar);

  size_t smem = (size_t)4 * 16 * LDW * sizeof(__bf16);  // ~97 KB (of 320 KB/WGP)
  lstm_persistent<<<NBLK, NTHR, smem, stream>>>(x, bf, bi, bc, bo, blin,
                                                Wg, Wl, comb, bar, out);
}
